// MLA_79551384256986
// MI455X (gfx1250) — matrix operand order repaired and checked by reading, not compiled, not run on hardware
//
#include <hip/hip_runtime.h>

// ---------------- constants (match reference) ----------------
#define Bb      2
#define Ss      2048
#define Hh      16
#define NOPE    128
#define ROPE    64
#define VDIM    128
#define QRANK   1536
#define KVRANK  512
#define QKD     (NOPE + ROPE)          // 192
#define SCALE   (0.07216878364870323f) // 1/sqrt(192)
#define NEGINF  (-10000.0f)

typedef __bf16 v16bf __attribute__((ext_vector_type(16)));
typedef float  v8f   __attribute__((ext_vector_type(8)));
typedef unsigned int u32x4 __attribute__((ext_vector_type(4)));
typedef int          i32x4 __attribute__((ext_vector_type(4)));
typedef int          i32x8 __attribute__((ext_vector_type(8)));

#if defined(__HIP_DEVICE_COMPILE__) && __has_builtin(__builtin_amdgcn_tensor_load_to_lds)
#define USE_TDM 1
#else
#define USE_TDM 0
#endif

union FragU { v16bf v; uint4 q[2]; };

static __device__ inline v8f wmma_bf16(v16bf a, v16bf b, v8f c) {
    return __builtin_amdgcn_wmma_f32_16x16x32_bf16(
        /*neg_a=*/false, a, /*neg_b=*/false, b,
        /*c_mod=*/(short)0, c, /*reuse_a=*/false, /*reuse_b=*/false);
}

// A-fragment (16x32 bf16): lane m = lane&15, half = lane>>4.
static __device__ inline v16bf ldfragA(const __bf16* rowp, int kc, int half) {
    FragU f;
    f.q[0] = *(const uint4*)(rowp + kc + 16 * half);
    f.q[1] = *(const uint4*)(rowp + kc + 16 * half + 8);
    return f.v;
}
// B-fragment (32x16 bf16): lane n = lane&15, khalf = lane>>4.
// Lane holds K = 16*khalf .. 16*khalf+15 contiguous (32 bytes).
static __device__ inline v16bf ldfragB(const __bf16* rowp, int off) {
    FragU f;
    f.q[0] = *(const uint4*)(rowp + off);
    f.q[1] = *(const uint4*)(rowp + off + 8);
    return f.v;
}

// ---------------- TDM descriptor builders (ISA cdna5 §8.3-8.6) ----------------
#if USE_TDM
// Group 0: count=1 [1:0], lds_addr [63:32], global_addr [120:64], type=2 [127:126]
static __device__ inline u32x4 tdm_g0(unsigned lds_byte, unsigned long long ga) {
    u32x4 g;
    g[0] = 1u;
    g[1] = lds_byte;
    g[2] = (unsigned)ga;
    g[3] = (unsigned)((ga >> 32) & 0x1FFFFFFu) | (2u << 30);
    return g;
}
// Group 1: mask=0 [15:0], data_size [17:16], tensor_dim0 [79:48],
// tensor_dim1 [111:80], tile_dim0 [127:112], tile_dim1 [143:128],
// tile_dim2 [159:144], td0_stride [207:160], td1_stride [255:208]
static __device__ inline i32x8 tdm_g1(unsigned td0, unsigned td1,
                                      unsigned tile0, unsigned tile1,
                                      unsigned tile2,
                                      unsigned long long st0) {
    i32x8 g;
    g[0] = (int)(1u << 16);  // data_size = 2 bytes
    g[1] = (int)((td0 & 0xFFFFu) << 16);
    g[2] = (int)((td0 >> 16) | ((td1 & 0xFFFFu) << 16));
    g[3] = (int)((td1 >> 16) | (tile0 << 16));
    g[4] = (int)(tile1 | (tile2 << 16));
    g[5] = (int)(unsigned)(st0 & 0xFFFFFFFFu);
    g[6] = (int)(unsigned)((st0 >> 32) & 0xFFFFu);  // td1_stride = 0
    g[7] = 0;
    return g;
}
#endif

// ---------------- stage 1: fp32 -> bf16 cast ----------------
__global__ void cast_f32_bf16(const float* __restrict__ in,
                              __bf16* __restrict__ out, int n) {
    for (int i = blockIdx.x * blockDim.x + threadIdx.x; i < n;
         i += gridDim.x * blockDim.x)
        out[i] = (__bf16)in[i];
}

// ---------------- stages 2&3: WMMA projection GEMM --------------
// C[M,N] = X[M,K] * W[N,K]^T   (all bf16, fp32 accum)
// mode 0 (Q):  N=3072, epilogue: split per head into nope / roped pe -> Qcat
// mode 1 (KV): N=4096, epilogue: k_nope -> Kcat, v -> vT (transposed)
__global__ __launch_bounds__(128)
void proj_gemm(const __bf16* __restrict__ X, const __bf16* __restrict__ W,
               int K, int mode,
               const float* __restrict__ cosp, const float* __restrict__ sinp,
               __bf16* __restrict__ qcat, __bf16* __restrict__ kcat,
               __bf16* __restrict__ vT) {
    __shared__ float cs[128 * 64];

    const int lane = threadIdx.x & 31;
    const int wave = threadIdx.x >> 5;
    const int ml   = lane & 15;
    const int half = lane >> 4;
    const int row0 = blockIdx.x * 128 + wave * 32;
    const int n0   = blockIdx.y * 64;

    const __bf16* xr0 = X + (size_t)(row0 + ml) * K;
    const __bf16* xr1 = X + (size_t)(row0 + 16 + ml) * K;

    v8f acc[2][4];
#pragma unroll
    for (int i = 0; i < 2; ++i)
#pragma unroll
        for (int j = 0; j < 4; ++j) acc[i][j] = (v8f)(0.0f);

    for (int kc = 0; kc < K; kc += 32) {
        if (kc + 64 < K) {  // speculative prefetch of next A chunk
            __builtin_prefetch(xr0 + kc + 64, 0, 1);
            __builtin_prefetch(xr1 + kc + 64, 0, 1);
        }
        v16bf a0 = ldfragA(xr0, kc, half);
        v16bf a1 = ldfragA(xr1, kc, half);
#pragma unroll
        for (int ns = 0; ns < 4; ++ns) {
            const __bf16* wr = W + (size_t)(n0 + ns * 16 + ml) * K;
            v16bf bf = ldfragB(wr, kc + 16 * half);
            acc[0][ns] = wmma_bf16(a0, bf, acc[0][ns]);
            acc[1][ns] = wmma_bf16(a1, bf, acc[1][ns]);
        }
    }

    // stage C tile (128x64 fp32) in LDS
#pragma unroll
    for (int ms = 0; ms < 2; ++ms)
#pragma unroll
        for (int ns = 0; ns < 4; ++ns)
#pragma unroll
            for (int g = 0; g < 8; ++g) {
                int rl = wave * 32 + ms * 16 + g + 8 * half;
                int cl = ns * 16 + ml;
                cs[rl * 64 + cl] = acc[ms][ns][g];
            }
    __syncthreads();

    // epilogue: one thread owns one row (64 cols) of the tile
    const int rl   = threadIdx.x;
    const int grow = blockIdx.x * 128 + rl;
    const int b    = grow / Ss;
    const int s    = grow % Ss;
    const float* rowv = &cs[rl * 64];

    if (mode == 0) {  // Q: cols over H*192
        for (int cl = 0; cl < 64; ++cl) {
            int gcol = n0 + cl;
            int h = gcol / QKD, r = gcol % QKD;
            float v = rowv[cl];
            size_t dst = (((size_t)(b * Hh + h) * Ss) + s) * QKD + r;
            if (r < NOPE) {
                qcat[dst] = (__bf16)v;
            } else {
                int j = r - NOPE;  // cl == j (64-aligned pe block)
                float part = rowv[(j < 32) ? cl + 32 : cl - 32];
                float rot  = (j < 32) ? -part : part;
                size_t ci  = (size_t)(b * Ss + s) * ROPE + j;
                qcat[dst] = (__bf16)(v * cosp[ci] + rot * sinp[ci]);
            }
        }
    } else {  // KV: cols over H*256
        for (int cl = 0; cl < 64; ++cl) {
            int gcol = n0 + cl;
            int h = gcol / (NOPE + VDIM), r = gcol % (NOPE + VDIM);
            float v = rowv[cl];
            if (r < NOPE) {
                kcat[(((size_t)(b * Hh + h) * Ss) + s) * QKD + r] = (__bf16)v;
            } else {
                int d = r - NOPE;
                vT[((size_t)(b * Hh + h) * VDIM + d) * Ss + s] = (__bf16)v;
            }
        }
    }
}

// ---------------- stage 4: rope(PE) broadcast into Kcat ----------------
__global__ void rope_pe(const float* __restrict__ PE,
                        const float* __restrict__ cosp,
                        const float* __restrict__ sinp,
                        __bf16* __restrict__ kcat) {
    int idx = blockIdx.x * blockDim.x + threadIdx.x;
    if (idx >= Bb * Ss * ROPE) return;
    int j = idx % ROPE;
    int s = (idx / ROPE) % Ss;
    int b = idx / (ROPE * Ss);
    float x = PE[idx];
    float part = PE[((size_t)b * Ss + s) * ROPE + ((j < 32) ? j + 32 : j - 32)];
    float rot  = (j < 32) ? -part : part;
    size_t ci = (size_t)(b * Ss + s) * ROPE + j;
    __bf16 v = (__bf16)(x * cosp[ci] + rot * sinp[ci]);
#pragma unroll
    for (int h = 0; h < Hh; ++h)
        kcat[(((size_t)(b * Hh + h) * Ss) + s) * QKD + NOPE + j] = v;
}

// ---------------- stage 5: flash attention (WMMA + TDM staging) ----------------
// grid.x = B*H, grid.y = S/64; 4 waves/block, 1 wave owns 16 q rows.
// K/V tiles of 32 k-positions are DMA'd into LDS (double-buffered) by the
// Tensor Data Mover; all waves consume them as B-fragments from LDS.
__global__ __launch_bounds__(128)
void mla_attn(const __bf16* __restrict__ qcat, const __bf16* __restrict__ kcat,
              const __bf16* __restrict__ vT, float* __restrict__ out) {
    __shared__ __align__(128) __bf16 ktile[2][32][QKD];   // 2 x 12 KB
    __shared__ __align__(128) __bf16 vtile[2][VDIM][32];  // 2 x 8 KB
    __shared__ __align__(16)  __bf16 pbuf[4][16][32];     // 4 KB

    const int bh   = blockIdx.x;
    const int b    = bh / Hh;
    const int h    = bh % Hh;
    const int lane = threadIdx.x & 31;
    const int wave = threadIdx.x >> 5;
    const int ml   = lane & 15;
    const int half = lane >> 4;
    const int qB   = blockIdx.y * 64;
    const int q0   = qB + wave * 16;

    const __bf16* Qb = qcat + (size_t)bh * Ss * QKD;
    const __bf16* Kb = kcat + (size_t)bh * Ss * QKD;
    const __bf16* Vb = vT   + (size_t)bh * VDIM * Ss;

    // preload the wave's Q A-fragments (16 rows x 192 = 6 K-chunks)
    v16bf aq[6];
    {
        const __bf16* qr = Qb + (size_t)(q0 + ml) * QKD;
#pragma unroll
        for (int c = 0; c < 6; ++c) aq[c] = ldfragA(qr, c * 32, half);
    }

    v8f o[8];
#pragma unroll
    for (int t = 0; t < 8; ++t) o[t] = (v8f)(0.0f);
    float Mrow[8], Lrow[8];
#pragma unroll
    for (int g = 0; g < 8; ++g) { Mrow[g] = -3.0e38f; Lrow[g] = 0.0f; }

    __bf16(*myp)[32] = pbuf[wave];
    const int kend = qB + 64;  // block-uniform causal bound; mask does the rest

#if USE_TDM
    // wave 0 drives the TDM; tiles pipelined one step ahead (TENSORcnt)
    if (wave == 0) {
        unsigned klds = (unsigned)(size_t)(void*)&ktile[0][0][0];
        unsigned vlds = (unsigned)(size_t)(void*)&vtile[0][0][0];
        // K tile: 32 contiguous rows of 192 bf16 -> 1-D tile of 6144 elems
        __builtin_amdgcn_tensor_load_to_lds(
            tdm_g0(klds, (unsigned long long)(size_t)Kb),
            tdm_g1(32 * QKD, 1, 32 * QKD, 1, 0, 32 * QKD),
            (i32x4)0, (i32x4)0, (i32x8)0, 0);
        // V tile: 128 rows x 32 elems, row stride S
        __builtin_amdgcn_tensor_load_to_lds(
            tdm_g0(vlds, (unsigned long long)(size_t)Vb),
            tdm_g1(Ss, VDIM, 32, VDIM, 1, Ss),
            (i32x4)0, (i32x4)0, (i32x8)0, 0);
    }
#endif

    int buf = 0;
    for (int kt = 0; kt < kend; kt += 32, buf ^= 1) {
#if USE_TDM
        if (wave == 0) __builtin_amdgcn_s_wait_tensorcnt(0);
        __syncthreads();  // current buffer filled; previous buffer free
        if (wave == 0 && kt + 32 < kend) {
            int nb = buf ^ 1;
            unsigned klds = (unsigned)(size_t)(void*)&ktile[nb][0][0];
            unsigned vlds = (unsigned)(size_t)(void*)&vtile[nb][0][0];
            __builtin_amdgcn_tensor_load_to_lds(
                tdm_g0(klds, (unsigned long long)(size_t)(Kb + (size_t)(kt + 32) * QKD)),
                tdm_g1(32 * QKD, 1, 32 * QKD, 1, 0, 32 * QKD),
                (i32x4)0, (i32x4)0, (i32x8)0, 0);
            __builtin_amdgcn_tensor_load_to_lds(
                tdm_g0(vlds, (unsigned long long)(size_t)(Vb + kt + 32)),
                tdm_g1(Ss, VDIM, 32, VDIM, 1, Ss),
                (i32x4)0, (i32x4)0, (i32x8)0, 0);
        }
#else
        __syncthreads();  // previous readers done before overwrite
        {   // cooperative global -> LDS staging
            const uint4* sk = (const uint4*)(Kb + (size_t)kt * QKD);
            uint4* dk = (uint4*)&ktile[buf][0][0];
            for (int i = threadIdx.x; i < (32 * QKD) / 8; i += 128) dk[i] = sk[i];
            const __bf16* gv = Vb + kt;
            for (int i = threadIdx.x; i < VDIM * 4; i += 128) {
                int d = i >> 2, c = (i & 3) * 8;
                *(uint4*)&vtile[buf][d][c] = *(const uint4*)(gv + (size_t)d * Ss + c);
            }
        }
        __syncthreads();
#endif

        // ---- scores: two 16x16 subtiles over K-dim 192 ----
        v8f sc[2];
#pragma unroll
        for (int sub = 0; sub < 2; ++sub) {
            v8f c = (v8f)(0.0f);
            const __bf16* kr = &ktile[buf][sub * 16 + ml][0];
#pragma unroll
            for (int ck = 0; ck < 6; ++ck)
                c = wmma_bf16(aq[ck], ldfragB(kr, ck * 32 + 16 * half), c);
            int kg = kt + sub * 16 + ml;  // C-layout: n = lane&15
#pragma unroll
            for (int g = 0; g < 8; ++g) {
                int qg = q0 + g + 8 * half;
                c[g] = c[g] * SCALE + ((kg > qg) ? NEGINF : 0.0f);
            }
            sc[sub] = c;
        }
        // ---- online softmax statistics (rows split by lane half) ----
        float alpha[8];
#pragma unroll
        for (int g = 0; g < 8; ++g) {
            float mx = fmaxf(sc[0][g], sc[1][g]);
#pragma unroll
            for (int msk = 1; msk < 16; msk <<= 1)
                mx = fmaxf(mx, __shfl_xor(mx, msk, 32));
            float nm = fmaxf(Mrow[g], mx);
            alpha[g] = __expf(Mrow[g] - nm);
            Mrow[g]  = nm;
            float p0 = __expf(sc[0][g] - nm);
            float p1 = __expf(sc[1][g] - nm);
            myp[g + 8 * half][ml]      = (__bf16)p0;
            myp[g + 8 * half][16 + ml] = (__bf16)p1;
            float sum = p0 + p1;
#pragma unroll
            for (int msk = 1; msk < 16; msk <<= 1)
                sum += __shfl_xor(sum, msk, 32);
            Lrow[g] = Lrow[g] * alpha[g] + sum;
        }
        // rescale accumulators
#pragma unroll
        for (int t = 0; t < 8; ++t)
#pragma unroll
            for (int g = 0; g < 8; ++g) o[t][g] *= alpha[g];

        // re-read probabilities as an A-fragment (per-wave LDS region;
        // wave-level LDS ops are in-order, fence only the compiler)
        asm volatile("" ::: "memory");
        FragU pa;
        pa.q[0] = *(const uint4*)&myp[ml][16 * half];
        pa.q[1] = *(const uint4*)&myp[ml][16 * half + 8];
        asm volatile("" ::: "memory");

        // ---- PV: 8 d-tiles of 16, K=32 ----
#pragma unroll
        for (int t = 0; t < 8; ++t) {
            const __bf16* vr = &vtile[buf][t * 16 + ml][0];
            o[t] = wmma_bf16(pa.v, ldfragB(vr, 16 * half), o[t]);
        }
    }

    // ---- epilogue: out[b, q, h, d] fp32 ----
#pragma unroll
    for (int t = 0; t < 8; ++t)
#pragma unroll
        for (int g = 0; g < 8; ++g) {
            int q = q0 + g + 8 * half;
            int d = t * 16 + ml;
            out[(((size_t)b * Ss + q) * Hh + h) * VDIM + d] = o[t][g] / Lrow[g];
        }
}

// ---------------- launcher ----------------
static inline size_t alignup(size_t x) { return (x + 255) & ~(size_t)255; }

extern "C" void kernel_launch(void* const* d_in, const int* in_sizes, int n_in,
                              void* d_out, int out_size, void* d_ws, size_t ws_size,
                              hipStream_t stream) {
    const float* Q    = (const float*)d_in[0];
    const float* KV   = (const float*)d_in[1];
    const float* PE   = (const float*)d_in[2];
    const float* WUQ  = (const float*)d_in[3];
    const float* WUKV = (const float*)d_in[4];
    const float* cosp = (const float*)d_in[5];
    const float* sinp = (const float*)d_in[6];
    float* out = (float*)d_out;

    const size_t nQ    = (size_t)Bb * Ss * QRANK;             // 6291456
    const size_t nKV   = (size_t)Bb * Ss * KVRANK;            // 2097152
    const size_t nWUQ  = (size_t)Hh * QKD * QRANK;            // 4718592
    const size_t nWUKV = (size_t)Hh * (NOPE + VDIM) * KVRANK; // 2097152
    const size_t nQcat = (size_t)Bb * Hh * Ss * QKD;          // 12582912
    const size_t nKcat = nQcat;
    const size_t nVT   = (size_t)Bb * Hh * VDIM * Ss;         // 8388608

    char* w = (char*)d_ws;
    __bf16* Qb    = (__bf16*)w;              w += alignup(nQ * 2);
    __bf16* KVb   = (__bf16*)w;              w += alignup(nKV * 2);
    __bf16* WUQb  = (__bf16*)w;              w += alignup(nWUQ * 2);
    __bf16* WUKVb = (__bf16*)w;              w += alignup(nWUKV * 2);
    __bf16* Qcat  = (__bf16*)w;              w += alignup(nQcat * 2);
    __bf16* Kcat  = (__bf16*)w;              w += alignup(nKcat * 2);
    __bf16* vT    = (__bf16*)w;              w += alignup(nVT * 2);

    // 1) casts
    cast_f32_bf16<<<2048, 256, 0, stream>>>(Q, Qb, (int)nQ);
    cast_f32_bf16<<<2048, 256, 0, stream>>>(KV, KVb, (int)nKV);
    cast_f32_bf16<<<2048, 256, 0, stream>>>(WUQ, WUQb, (int)nWUQ);
    cast_f32_bf16<<<2048, 256, 0, stream>>>(WUKV, WUKVb, (int)nWUKV);

    // 2) q = Q*WUQ^T  (M=4096, N=3072, K=1536) + rope -> Qcat
    proj_gemm<<<dim3(32, 48), 128, 0, stream>>>(Qb, WUQb, QRANK, 0,
                                                cosp, sinp, Qcat, nullptr, nullptr);
    // 3) kv = KV*WUKV^T (M=4096, N=4096, K=512) -> Kcat / vT
    proj_gemm<<<dim3(32, 64), 128, 0, stream>>>(KVb, WUKVb, KVRANK, 1,
                                                cosp, sinp, nullptr, Kcat, vT);
    // 4) k_pe = rope(PE) -> Kcat[..., 128:192] for all heads
    rope_pe<<<(Bb * Ss * ROPE + 255) / 256, 256, 0, stream>>>(PE, cosp, sinp, Kcat);

    // 5) causal flash attention with TDM-staged K/V tiles
    mla_attn<<<dim3(Bb * Hh, Ss / 64), 128, 0, stream>>>(Qcat, Kcat, vT, out);

    (void)in_sizes; (void)n_in; (void)out_size; (void)ws_size;
}
